// SimpleSmoothParticleNet_18915035972384
// MI455X (gfx1250) — matrix operand order repaired and checked by reading, not compiled, not run on hardware
//
#include <hip/hip_runtime.h>
#include <hip/hip_bf16.h>

// ---------------------------------------------------------------------------
// CDNA5 (gfx1250) implementation of SimpleSmoothParticleNet.
// 32->32 conv layers: implicit GEMM on v_wmma_f32_16x16x32_f16 (f16 in, f32
// accum), M=32 voxels x N=32 channels per wave (4 accumulators / 2 A-frags /
// 2 B-frags per tap). Weights are pre-packed to the WMMA B-matrix f16 layout
// once per launch; conv blocks stage them into LDS with
// GLOBAL_LOAD_ASYNC_TO_LDS_B128 (ASYNCcnt) when the builtin is available.
// ---------------------------------------------------------------------------

typedef __attribute__((ext_vector_type(16))) _Float16 v16h;
typedef __attribute__((ext_vector_type(8)))  float    v8f;

struct alignas(16) U128 { unsigned int x, y, z, w; };
struct alignas(16) F128 { float x, y, z, w; };

union Frag16 {            // one WMMA 16-bit operand: 8 VGPRs = 16 halves
    v16h v;
    U128 q[2];
};

#define GD   64
#define NVOX (2L * GD * GD * GD)
#define WPACK_DWORDS (27 * 2 * 32 * 8)   // 13824 dwords = 55296 B per layer

#if defined(__HIP_DEVICE_COMPILE__) && __has_builtin(__builtin_amdgcn_global_load_async_to_lds_b128)
#define USE_ASYNC_LDS 1
#else
#define USE_ASYNC_LDS 0
#endif

#if USE_ASYNC_LDS
// clang-22 declares the async-to-LDS builtins with typed vector pointers:
//   void __builtin_amdgcn_global_load_async_to_lds_b128(v4i AS1*, v4i AS3*, Ii, Ii)
typedef int v4i __attribute__((vector_size(16)));
typedef __attribute__((address_space(1))) v4i gv4i;
typedef __attribute__((address_space(3))) v4i lv4i;

__device__ __forceinline__ void async_copy16(const void* g, void* l) {
    // global -> LDS 16B async copy (tracked by ASYNCcnt)
    __builtin_amdgcn_global_load_async_to_lds_b128(
        (gv4i*)(unsigned long long)g,
        (lv4i*)(unsigned int)(unsigned long long)l,
        0, 0);
}
__device__ __forceinline__ void wait_async0() {
#if __has_builtin(__builtin_amdgcn_s_wait_asynccnt)
    __builtin_amdgcn_s_wait_asynccnt(0);
#else
    asm volatile("s_wait_asynccnt 0" ::: "memory");
#endif
}
#endif

__device__ __forceinline__ long voxIdx(int b, int x, int y, int z) {
    return (((long)b * GD + x) * GD + y) * GD + z;
}

// ---------------------------------------------------------------------------
// zero-fill f32 buffer
// ---------------------------------------------------------------------------
__global__ void zero_f32(float* __restrict__ p, long n) {
    long i = (long)blockIdx.x * blockDim.x + threadIdx.x;
    if (i < n) p[i] = 0.0f;
}

// ---------------------------------------------------------------------------
// Pre-pack one 32x32x27 f32 weight tensor into WMMA B-fragment f16 layout:
//   wpack[((t*2 + nh)*32 + lane)*8 + j] = pair(k0, k0+1) for
//   n = nh*16 + (lane&15), k0 = 16*(lane>>4) + 2*j.
// ---------------------------------------------------------------------------
__global__ __launch_bounds__(256) void pack_weights_32x32(
    const float* __restrict__ W, unsigned int* __restrict__ wpack)
{
    const int idx = blockIdx.x * 256 + threadIdx.x;     // 13824 total, exact grid
    const int j  = idx & 7;
    const int L  = (idx >> 3) & 31;
    const int nh = (idx >> 8) & 1;
    const int t  = idx >> 9;
    const int n  = nh * 16 + (L & 15);
    const int k0 = 16 * (L >> 4) + 2 * j;
    _Float16 w0 = (_Float16)W[(n * 32 + k0)     * 27 + t];   // W is [32][32][27]
    _Float16 w1 = (_Float16)W[(n * 32 + k0 + 1) * 27 + t];
    wpack[idx] = (unsigned int)*(unsigned short*)&w0 |
                 ((unsigned int)*(unsigned short*)&w1 << 16);
}

// ---------------------------------------------------------------------------
// particles2grid: poly6 splat, one thread per particle, atomicAdd 4 channels
// ---------------------------------------------------------------------------
__global__ __launch_bounds__(256) void splat_kernel(
    const float* __restrict__ locs, const float* __restrict__ data,
    const float* __restrict__ density, float* __restrict__ grid)
{
    const int pid = blockIdx.x * 256 + threadIdx.x;      // 0 .. B*N-1
    const int b   = pid >> 15;                           // N = 32768
    const float lx = locs[pid * 3 + 0];
    const float ly = locs[pid * 3 + 1];
    const float lz = locs[pid * 3 + 2];
    const float inv_step = 64.0f;                        // 1/0.015625
    const float step     = 0.015625f;
    const int bx = (int)floorf(lx * inv_step);
    const int by = (int)floorf(ly * inv_step);
    const int bz = (int)floorf(lz * inv_step);
    const float ds = 1.0f / density[pid];
    const float d0 = data[pid * 4 + 0] * ds;
    const float d1 = data[pid * 4 + 1] * ds;
    const float d2c = data[pid * 4 + 2] * ds;
    const float d3 = data[pid * 4 + 3] * ds;
    const float h2 = 0.0009765625f;                      // (0.03125)^2
    // 315/(64*pi*h^9) = 315/(64*pi) * 2^45
    const float C6 = (315.0f / (64.0f * 3.14159265358979f)) * 35184372088832.0f;

    for (int ox = -2; ox <= 2; ++ox)
      for (int oy = -2; oy <= 2; ++oy)
        for (int oz = -2; oz <= 2; ++oz) {
            const int cx = bx + ox, cy = by + oy, cz = bz + oz;
            if ((unsigned)cx < 64u && (unsigned)cy < 64u && (unsigned)cz < 64u) {
                const float ex = ((float)cx + 0.5f) * step - lx;
                const float ey = ((float)cy + 0.5f) * step - ly;
                const float ez = ((float)cz + 0.5f) * step - lz;
                const float r2 = ex * ex + ey * ey + ez * ez;
                if (r2 <= h2) {
                    const float u = h2 - r2;
                    const float w = C6 * u * u * u;
                    float* g = grid + voxIdx(b, cx, cy, cz) * 4;
                    atomicAdd(g + 0, w * d0);
                    atomicAdd(g + 1, w * d1);
                    atomicAdd(g + 2, w * d2c);
                    atomicAdd(g + 3, w * d3);
                }
            }
        }
}

// ---------------------------------------------------------------------------
// conv0: 4 -> 32 channels, PReLU, f32 in -> f16 out. One thread per voxel.
// ---------------------------------------------------------------------------
__global__ __launch_bounds__(256) void conv0_4to32(
    const float* __restrict__ in, _Float16* __restrict__ out,
    const float* __restrict__ W, const float* __restrict__ bias,
    const float* __restrict__ alpha)
{
    __shared__ float wlds[27 * 4 * 32];          // [t][c][n]
    __shared__ float blds[32];
    for (int idx = threadIdx.x; idx < 27 * 4 * 32; idx += 256) {
        const int n = idx & 31, c = (idx >> 5) & 3, t = idx >> 7;
        wlds[idx] = W[(n * 4 + c) * 27 + t];     // W0 is [32][4][3][3][3]
    }
    if (threadIdx.x < 32) blds[threadIdx.x] = bias[threadIdx.x];
    __syncthreads();

    const int vid = blockIdx.x * 256 + threadIdx.x;    // 0 .. NVOX-1
    const int z = vid & 63, yv = (vid >> 6) & 63, xv = (vid >> 12) & 63, b = vid >> 18;

    float acc[32];
#pragma unroll
    for (int n = 0; n < 32; ++n) acc[n] = 0.0f;

    const float* pin = in + voxIdx(b, xv, yv, z) * 4;
#pragma unroll
    for (int t = 0; t < 27; ++t) {
        const int dz = t % 3 - 1, dy = (t / 3) % 3 - 1, dx = t / 9 - 1;
        const bool vxy = ((unsigned)(xv + dx) < 64u) && ((unsigned)(yv + dy) < 64u);
        if (vxy && (unsigned)(z + dz) < 64u) {
            const long off = (((long)dx * GD + dy) * GD + dz) * 4;
            const F128 v = *reinterpret_cast<const F128*>(pin + off);
            const float* wt = &wlds[t * 128];
#pragma unroll
            for (int n = 0; n < 32; ++n)
                acc[n] += wt[n] * v.x + wt[32 + n] * v.y + wt[64 + n] * v.z + wt[96 + n] * v.w;
        }
    }

    const float al = alpha[0];
    unsigned int packed[16];
#pragma unroll
    for (int i = 0; i < 16; ++i) {
        float v0 = acc[2 * i] + blds[2 * i];
        float v1 = acc[2 * i + 1] + blds[2 * i + 1];
        v0 = v0 >= 0.0f ? v0 : al * v0;
        v1 = v1 >= 0.0f ? v1 : al * v1;
        _Float16 h0 = (_Float16)v0, h1 = (_Float16)v1;
        packed[i] = (unsigned int)*(unsigned short*)&h0 |
                    ((unsigned int)*(unsigned short*)&h1 << 16);
    }
    U128* op = reinterpret_cast<U128*>(out + voxIdx(b, xv, yv, z) * 32);
    op[0] = *reinterpret_cast<U128*>(&packed[0]);
    op[1] = *reinterpret_cast<U128*>(&packed[4]);
    op[2] = *reinterpret_cast<U128*>(&packed[8]);
    op[3] = *reinterpret_cast<U128*>(&packed[12]);
}

// ---------------------------------------------------------------------------
// WMMA implicit-GEMM conv: 32 -> 32 channels, PReLU, f16 -> f16.
// One wave computes 32 z-consecutive voxels x 32 output channels:
//   per tap: A0/A1 = 16(vox) x 32(cin) f16, B0/B1 = 32(cin) x 16(cout) f16,
//   4x v_wmma_f32_16x16x32_f16, f32 accumulators.
// Pre-packed B fragments are staged global->LDS with async-to-LDS copies.
// ---------------------------------------------------------------------------
__global__ __launch_bounds__(256) void conv_wmma_32x32(
    const _Float16* __restrict__ in, _Float16* __restrict__ out,
    const unsigned int* __restrict__ wpack, const float* __restrict__ bias,
    const float* __restrict__ alpha)
{
    __shared__ unsigned int wlds[WPACK_DWORDS];   // [tap][nhalf][lane][pairVGPR]

#if USE_ASYNC_LDS
    for (int idx = threadIdx.x; idx < WPACK_DWORDS / 4; idx += 256)
        async_copy16(wpack + idx * 4, &wlds[idx * 4]);
    wait_async0();
#else
    for (int idx = threadIdx.x; idx < WPACK_DWORDS / 4; idx += 256)
        *reinterpret_cast<U128*>(&wlds[idx * 4]) =
            reinterpret_cast<const U128*>(wpack)[idx];
#endif
    __syncthreads();

    const int lane = threadIdx.x & 31;
    const int wave = threadIdx.x >> 5;
    const int tile = blockIdx.x * 8 + wave;          // 16384 tiles total (exact)
    const int z0 = (tile & 1) * 32;
    const int y  = (tile >> 1) & 63;
    const int x  = (tile >> 7) & 63;
    const int b  = (tile >> 13) & 1;

    const int m  = lane & 15;                        // A-matrix row = voxel in tile
    const int hh = lane >> 4;
    const int zm = z0 + m;

    // per-lane A base pointers (chunk1 is +8 channels from chunk0)
    const _Float16* pA0 = in + voxIdx(b, x, y, zm) * 32 + 16 * hh;
    const _Float16* pA1 = pA0 + 16 * 32;             // +16 voxels in z
    const unsigned int* wl = wlds + lane * 8;        // per-lane B fragment base

    v8f acc00 = {0.f, 0.f, 0.f, 0.f, 0.f, 0.f, 0.f, 0.f};
    v8f acc01 = acc00, acc10 = acc00, acc11 = acc00;

#pragma unroll
    for (int t = 0; t < 27; ++t) {
        const int dz = t % 3 - 1, dy = (t / 3) % 3 - 1, dx = t / 9 - 1;
        // (x,y) validity is wave-uniform -> scalar; z validity is per-lane
        const bool vxy = ((unsigned)(x + dx) < 64u) && ((unsigned)(y + dy) < 64u);
        const long off = (((long)dx * GD + dy) * GD + dz) * 32;

        Frag16 a0, a1;
        U128 zq = {0u, 0u, 0u, 0u};
        if (vxy && (unsigned)(zm + dz) < 64u) {
            a0.q[0] = *reinterpret_cast<const U128*>(pA0 + off);
            a0.q[1] = *reinterpret_cast<const U128*>(pA0 + off + 8);
        } else { a0.q[0] = zq; a0.q[1] = zq; }
        if (vxy && (unsigned)(zm + 16 + dz) < 64u) {
            a1.q[0] = *reinterpret_cast<const U128*>(pA1 + off);
            a1.q[1] = *reinterpret_cast<const U128*>(pA1 + off + 8);
        } else { a1.q[0] = zq; a1.q[1] = zq; }

        Frag16 bf0, bf1;
        const U128* wp0 = reinterpret_cast<const U128*>(wl + t * 512);
        const U128* wp1 = reinterpret_cast<const U128*>(wl + t * 512 + 256);
        bf0.q[0] = wp0[0]; bf0.q[1] = wp0[1];
        bf1.q[0] = wp1[0]; bf1.q[1] = wp1[1];

        acc00 = __builtin_amdgcn_wmma_f32_16x16x32_f16(false, a0.v, false, bf0.v,
                                                       (short)0, acc00, false, false);
        acc01 = __builtin_amdgcn_wmma_f32_16x16x32_f16(false, a0.v, false, bf1.v,
                                                       (short)0, acc01, false, false);
        acc10 = __builtin_amdgcn_wmma_f32_16x16x32_f16(false, a1.v, false, bf0.v,
                                                       (short)0, acc10, false, false);
        acc11 = __builtin_amdgcn_wmma_f32_16x16x32_f16(false, a1.v, false, bf1.v,
                                                       (short)0, acc11, false, false);
    }

    // Epilogue: bias + PReLU, f16 channel-last store.
    // C/D layout: lane -> column n = lane&15; VGPR r -> row m = r + 8*(lane>>4).
    const float al  = alpha[0];
    const int   n0  = lane & 15;
    const float bi0 = bias[n0];
    const float bi1 = bias[16 + n0];
    _Float16* ob0 = out + voxIdx(b, x, y, z0 + 8 * hh) * 32 + n0;
    _Float16* ob1 = ob0 + 16 * 32;
#pragma unroll
    for (int r = 0; r < 8; ++r) {
        float v00 = acc00[r] + bi0, v01 = acc01[r] + bi1;
        float v10 = acc10[r] + bi0, v11 = acc11[r] + bi1;
        v00 = v00 >= 0.0f ? v00 : al * v00;
        v01 = v01 >= 0.0f ? v01 : al * v01;
        v10 = v10 >= 0.0f ? v10 : al * v10;
        v11 = v11 >= 0.0f ? v11 : al * v11;
        ob0[r * 32]      = (_Float16)v00;
        ob0[r * 32 + 16] = (_Float16)v01;
        ob1[r * 32]      = (_Float16)v10;
        ob1[r * 32 + 16] = (_Float16)v11;
    }
}

// ---------------------------------------------------------------------------
// conv3: 32 -> 4 channels, bias only, f16 in -> f32 out. One thread per voxel.
// ---------------------------------------------------------------------------
__global__ __launch_bounds__(256) void conv3_32to4(
    const _Float16* __restrict__ in, float* __restrict__ out,
    const float* __restrict__ W, const float* __restrict__ bias)
{
    __shared__ float wlds[27 * 32 * 4];          // [t][c][n]
    for (int idx = threadIdx.x; idx < 27 * 32 * 4; idx += 256) {
        const int n = idx & 3, c = (idx >> 2) & 31, t = idx >> 7;
        wlds[idx] = W[(n * 32 + c) * 27 + t];    // W3 is [4][32][3][3][3]
    }
    __syncthreads();

    const int vid = blockIdx.x * 256 + threadIdx.x;
    const int z = vid & 63, yv = (vid >> 6) & 63, xv = (vid >> 12) & 63, b = vid >> 18;

    float a0 = 0.f, a1 = 0.f, a2 = 0.f, a3 = 0.f;
    union H32 { U128 q[4]; _Float16 h[32]; };

    const _Float16* pin = in + voxIdx(b, xv, yv, z) * 32;
#pragma unroll
    for (int t = 0; t < 27; ++t) {
        const int dz = t % 3 - 1, dy = (t / 3) % 3 - 1, dx = t / 9 - 1;
        const bool vxy = ((unsigned)(xv + dx) < 64u) && ((unsigned)(yv + dy) < 64u);
        if (vxy && (unsigned)(z + dz) < 64u) {
            const long off = (((long)dx * GD + dy) * GD + dz) * 32;
            const U128* p = reinterpret_cast<const U128*>(pin + off);
            H32 buf;
            buf.q[0] = p[0]; buf.q[1] = p[1]; buf.q[2] = p[2]; buf.q[3] = p[3];
            const float* wt = &wlds[t * 128];
#pragma unroll
            for (int c = 0; c < 32; ++c) {
                const float v = (float)buf.h[c];
                a0 += wt[c * 4 + 0] * v;
                a1 += wt[c * 4 + 1] * v;
                a2 += wt[c * 4 + 2] * v;
                a3 += wt[c * 4 + 3] * v;
            }
        }
    }
    F128 o;
    o.x = a0 + bias[0]; o.y = a1 + bias[1]; o.z = a2 + bias[2]; o.w = a3 + bias[3];
    *reinterpret_cast<F128*>(out + voxIdx(b, xv, yv, z) * 4) = o;
}

// ---------------------------------------------------------------------------
// grid2particles: trilinear gather, one thread per particle
// ---------------------------------------------------------------------------
__global__ __launch_bounds__(256) void gather_trilinear(
    const float* __restrict__ grid, const float* __restrict__ locs,
    float* __restrict__ out)
{
    const int pid = blockIdx.x * 256 + threadIdx.x;
    const int b   = pid >> 15;
    const float px = locs[pid * 3 + 0] * 64.0f - 0.5f;
    const float py = locs[pid * 3 + 1] * 64.0f - 0.5f;
    const float pz = locs[pid * 3 + 2] * 64.0f - 0.5f;
    const int ix = (int)floorf(px), iy = (int)floorf(py), iz = (int)floorf(pz);
    const float fx = px - (float)ix, fy = py - (float)iy, fz = pz - (float)iz;

    float o0 = 0.f, o1 = 0.f, o2 = 0.f, o3 = 0.f;
#pragma unroll
    for (int dx = 0; dx < 2; ++dx)
#pragma unroll
      for (int dy = 0; dy < 2; ++dy)
#pragma unroll
        for (int dz = 0; dz < 2; ++dz) {
            const int cx = ix + dx, cy = iy + dy, cz = iz + dz;
            if ((unsigned)cx < 64u && (unsigned)cy < 64u && (unsigned)cz < 64u) {
                const float w = (dx ? fx : 1.0f - fx) * (dy ? fy : 1.0f - fy)
                              * (dz ? fz : 1.0f - fz);
                const F128 g = *reinterpret_cast<const F128*>(
                    grid + voxIdx(b, cx, cy, cz) * 4);
                o0 += w * g.x; o1 += w * g.y; o2 += w * g.z; o3 += w * g.w;
            }
        }
    F128 o; o.x = o0; o.y = o1; o.z = o2; o.w = o3;
    *reinterpret_cast<F128*>(out + (long)pid * 4) = o;
}

// ---------------------------------------------------------------------------
extern "C" void kernel_launch(void* const* d_in, const int* in_sizes, int n_in,
                              void* d_out, int out_size, void* d_ws, size_t ws_size,
                              hipStream_t stream)
{
    const float* locs = (const float*)d_in[0];
    const float* data = (const float*)d_in[1];
    const float* dens = (const float*)d_in[2];
    const float* W0   = (const float*)d_in[3];
    const float* b0   = (const float*)d_in[4];
    const float* W1   = (const float*)d_in[5];
    const float* b1   = (const float*)d_in[6];
    const float* W2   = (const float*)d_in[7];
    const float* b2   = (const float*)d_in[8];
    const float* W3   = (const float*)d_in[9];
    const float* b3   = (const float*)d_in[10];
    const float* a0   = (const float*)d_in[11];
    const float* a1   = (const float*)d_in[12];
    const float* a2   = (const float*)d_in[13];

    char* ws = (char*)d_ws;
    float*        grid0  = (float*)ws;                       //  8 MB f32 [2,64^3,4]
    _Float16*     act1   = (_Float16*)(ws + (8u  << 20));    // 32 MB f16 [2,64^3,32]
    _Float16*     act2   = (_Float16*)(ws + (40u << 20));    // 32 MB f16 [2,64^3,32]
    unsigned int* wpack1 = (unsigned int*)(ws + (72u << 20));            // 55296 B
    unsigned int* wpack2 = (unsigned int*)(ws + (72u << 20) + (64u << 10));
    float*        out    = (float*)d_out;                    // [2,32768,4]

    const int NP = 2 * 32768;

    zero_f32<<<(int)((NVOX * 4 + 255) / 256), 256, 0, stream>>>(grid0, NVOX * 4);
    pack_weights_32x32<<<WPACK_DWORDS / 256, 256, 0, stream>>>(W1, wpack1);
    pack_weights_32x32<<<WPACK_DWORDS / 256, 256, 0, stream>>>(W2, wpack2);
    splat_kernel<<<NP / 256, 256, 0, stream>>>(locs, data, dens, grid0);
    conv0_4to32<<<(int)(NVOX / 256), 256, 0, stream>>>(grid0, act1, W0, b0, a0);
    conv_wmma_32x32<<<2048, 256, 0, stream>>>(act1, act2, wpack1, b1, a1); // layer 1
    conv_wmma_32x32<<<2048, 256, 0, stream>>>(act2, act1, wpack2, b2, a2); // layer 2
    conv3_32to4<<<(int)(NVOX / 256), 256, 0, stream>>>(act1, grid0, W3, b3);
    gather_trilinear<<<NP / 256, 256, 0, stream>>>(grid0, locs, out);
}
